// GatedShortBlock_31387620999326
// MI455X (gfx1250) — matrix operand order repaired and checked by reading, not compiled, not run on hardware
//
#include <hip/hip_runtime.h>
#include <stdint.h>

// ---------------- types ----------------
typedef __attribute__((ext_vector_type(16))) __bf16       v16bf;
typedef __attribute__((ext_vector_type(8)))  float        v8f;
typedef __attribute__((ext_vector_type(4)))  unsigned int uv4;

// ---------------- problem constants ----------------
constexpr int BATCH = 2, SEQ = 4096, DM = 2048;
constexpr int M_TOT = BATCH * SEQ;   // 8192 rows
constexpr int KDIM  = DM;            // 2048 reduction dim (both GEMMs)

// ---------------- GEMM tiling ----------------
constexpr int BM = 128, BN = 128, BK = 32;
constexpr int LDT = 40;              // padded LDS row stride in bf16 elems (80 B)

// ---------------- CDNA5 async global->LDS (ASYNCcnt path) ----------------
// LDS destination = low 32 bits of the generic pointer (flat aperture rule:
// offset is the address truncated to 32 bits); global address per-lane in a VGPR pair.
#if defined(__HIP_DEVICE_COMPILE__)
#define ASYNC_LOAD_B128(ldsoff, gptr)                                   \
  asm volatile("global_load_async_to_lds_b128 %0, %1, off"              \
               :: "v"(ldsoff), "v"(gptr) : "memory")
#define S_WAIT_ASYNCCNT(imm_str)                                        \
  asm volatile("s_wait_asynccnt " imm_str ::: "memory")
#else
#define ASYNC_LOAD_B128(ldsoff, gptr) ((void)(ldsoff), (void)(gptr))
#define S_WAIT_ASYNCCNT(imm_str) ((void)0)
#endif

union Frag {
  uv4   q[2];
  v16bf v;
};

__device__ __forceinline__ v8f wmma_bf16(v16bf a, v16bf b, v8f c) {
#if defined(__HIP_DEVICE_COMPILE__)
  return __builtin_amdgcn_wmma_f32_16x16x32_bf16(
      /*neg_a=*/false, a, /*neg_b=*/false, b,
      /*c_mod=*/(short)0, c, /*reuse_a=*/false, /*reuse_b=*/false);
#else
  (void)a; (void)b;
  return c;
#endif
}

// A fragment (16x32, MxK): lanes 0-15 hold M=lane, K=[0..8)+[16..24);
// lanes 16-31 hold M=lane-16, K=[8..16)+[24..32). Two 16B runs per lane.
__device__ __forceinline__ v16bf load_frag_a(const unsigned short* As, int row0, int lane) {
  const int m  = lane & 15;
  const int kb = (lane >> 4) * 8;
  const int base = (row0 + m) * LDT + kb;
  Frag f;
  f.q[0] = *reinterpret_cast<const uv4*>(As + base);        // K kb..kb+7
  f.q[1] = *reinterpret_cast<const uv4*>(As + base + 16);   // K kb+16..kb+23
  return f.v;
}

// B fragment (32x16, KxN) from row-major W (N rows of K): lanes 0-15 hold
// N=lane, K=[0..8)+[16..24); lanes 16-31 hold N=lane-16, K=[8..16)+[24..32). Two 16B runs.
__device__ __forceinline__ v16bf load_frag_b(const unsigned short* Bs, int row0, int lane) {
  const int n  = lane & 15;
  const int kb = (lane >> 4) * 8;
  const int base = (row0 + n) * LDT + kb;
  Frag f;
  f.q[0] = *reinterpret_cast<const uv4*>(Bs + base);        // K kb..kb+7
  f.q[1] = *reinterpret_cast<const uv4*>(Bs + base + 16);   // K kb+16..kb+23
  return f.v;
}

// Stage one BMxBK A-tile and NG BNxBK B-tiles into LDS via async loads.
// Per thread: 2 + 2*NG async b128 instructions (8 for NG=3, 4 for NG=1).
template <int NG>
__device__ __forceinline__ void stage_async(
    const __bf16* __restrict__ A, const __bf16* __restrict__ Bw,
    unsigned short* asbuf, unsigned short* bsbuf,
    int m0, int n0, int kt, int tid) {
#pragma unroll
  for (int c = 0; c < 2; ++c) {
    const int id  = tid + c * 256;
    const int row = id >> 2;
    const int kc  = (id & 3) * 8;
    const __bf16* gp = A + (size_t)(m0 + row) * KDIM + kt * BK + kc;
    const uint32_t lo = (uint32_t)(uintptr_t)(asbuf + row * LDT + kc);
    ASYNC_LOAD_B128(lo, gp);
  }
#pragma unroll
  for (int g = 0; g < NG; ++g) {
#pragma unroll
    for (int c = 0; c < 2; ++c) {
      const int id  = tid + c * 256;
      const int row = id >> 2;
      const int kc  = (id & 3) * 8;
      const __bf16* gp = Bw + (size_t)(g * DM + n0 + row) * KDIM + kt * BK + kc;
      const uint32_t lo = (uint32_t)(uintptr_t)(bsbuf + (g * BN + row) * LDT + kc);
      ASYNC_LOAD_B128(lo, gp);
    }
  }
}

// ---------------- fused bf16 GEMM (double-buffered async staging) ----------
// NG==3: A=xb [M,K], Bw=w1b [3D,K]; computes Bg/Cg/Xg tiles for the same
//        (m,n) range, epilogue writes u=Bg*Xg and Cg (both f32 [M,D]).
// NG==1: A=gated_b [M,K], Bw=w2b [D,K]; writes f32 out [M,D].
template <int NG>
__global__ __launch_bounds__(256, 1)
void gemm_bf16_wmma(const __bf16* __restrict__ A, const __bf16* __restrict__ Bw,
                    float* __restrict__ out0, float* __restrict__ out1) {
  __shared__ alignas(16) unsigned short As[2][BM * LDT];       // 2 x 10 KB
  __shared__ alignas(16) unsigned short Bs[2][NG][BN * LDT];   // 2 x NG x 10 KB

  const int tid  = threadIdx.x;
  const int lane = tid & 31;
  const int wave = tid >> 5;
  const int wm   = (wave >> 1) * 32;   // 4 waves along M
  const int wn   = (wave & 1) * 64;    // 2 waves along N

  const int n0 = blockIdx.x * BN;      // in [0, DM)
  const int m0 = blockIdx.y * BM;

  const v8f vzero = {};
  v8f acc[NG][2][4];
  for (int g = 0; g < NG; ++g)
    for (int i = 0; i < 2; ++i)
      for (int j = 0; j < 4; ++j)
        acc[g][i][j] = vzero;

  const int nsteps = KDIM / BK;

  // prologue: tile 0 -> buffer 0
  stage_async<NG>(A, Bw, &As[0][0], &Bs[0][0][0], m0, n0, 0, tid);

  for (int kt = 0; kt < nsteps; ++kt) {
    const int cur = kt & 1;

    // All waves finished reading buffer cur^1 (during iteration kt-1);
    // safe to overwrite it with tile kt+1.
    __syncthreads();
    if (kt + 1 < nsteps) {
      stage_async<NG>(A, Bw, &As[cur ^ 1][0], &Bs[cur ^ 1][0][0],
                      m0, n0, kt + 1, tid);
      // async ops complete in order: once <= (2+2*NG) remain, tile kt landed.
      if constexpr (NG == 3) { S_WAIT_ASYNCCNT("8"); }
      else                   { S_WAIT_ASYNCCNT("4"); }
    } else {
      S_WAIT_ASYNCCNT("0");
    }
    __syncthreads();  // tile kt resident in LDS for every wave

    // ---- compute: one 16x16x32 WMMA K-step per fragment pair ----
    v16bf afrag[2];
    afrag[0] = load_frag_a(&As[cur][0], wm, lane);
    afrag[1] = load_frag_a(&As[cur][0], wm + 16, lane);
#pragma unroll
    for (int g = 0; g < NG; ++g) {
#pragma unroll
      for (int fn = 0; fn < 4; ++fn) {
        const v16bf bfrag = load_frag_b(&Bs[cur][g][0], wn + fn * 16, lane);
#pragma unroll
        for (int fm = 0; fm < 2; ++fm)
          acc[g][fm][fn] = wmma_bf16(afrag[fm], bfrag, acc[g][fm][fn]);
      }
    }
  }

  // ---- epilogue ----
  // C/D layout: VGPR r <-> M = r + 8*(lane>=16), N = lane%16
  const int mrow0 = m0 + wm + (lane >> 4) * 8;
  const int ncol0 = n0 + wn + (lane & 15);
#pragma unroll
  for (int fm = 0; fm < 2; ++fm) {
#pragma unroll
    for (int fn = 0; fn < 4; ++fn) {
      const int ncol = ncol0 + fn * 16;
      if constexpr (NG == 3) {
        const v8f uv = acc[0][fm][fn] * acc[2][fm][fn];  // Bg * Xg
        const v8f cv = acc[1][fm][fn];                   // Cg
#pragma unroll
        for (int r = 0; r < 8; ++r) {
          const size_t idx = (size_t)(mrow0 + fm * 16 + r) * DM + ncol;
          out0[idx] = uv[r];
          out1[idx] = cv[r];
        }
      } else {
        const v8f ov = acc[0][fm][fn];
#pragma unroll
        for (int r = 0; r < 8; ++r) {
          const size_t idx = (size_t)(mrow0 + fm * 16 + r) * DM + ncol;
          out0[idx] = ov[r];
        }
      }
    }
  }
}

// ---------------- fp32 -> bf16 convert ----------------
__global__ void cvt_f32_bf16(const float* __restrict__ in,
                             __bf16* __restrict__ out, int n) {
  int i = blockIdx.x * blockDim.x + threadIdx.x;
  const int stride = gridDim.x * blockDim.x;
  for (; i < n; i += stride) out[i] = (__bf16)in[i];
}

// ---------------- causal depthwise conv (K=3) + gate ----------------
// gated[b,s,d] = Cg[b,s,d] * (w0*u[s-2] + w1*u[s-1] + w2*u[s]); bf16 out.
constexpr int SCHUNK = 64;
__global__ __launch_bounds__(256, 1)
void conv3_gate(const float* __restrict__ u, const float* __restrict__ cg,
                const float* __restrict__ cw, __bf16* __restrict__ gb) {
  const int d  = blockIdx.x * 256 + threadIdx.x;
  const int s0 = blockIdx.y * SCHUNK;
  const int b  = blockIdx.z;
  const float w0 = cw[d * 3 + 0];
  const float w1 = cw[d * 3 + 1];
  const float w2 = cw[d * 3 + 2];
  const size_t base = (size_t)b * SEQ * DM + d;
  float um1 = (s0 >= 1) ? u[base + (size_t)(s0 - 1) * DM] : 0.0f;
  float um2 = (s0 >= 2) ? u[base + (size_t)(s0 - 2) * DM] : 0.0f;
  for (int s = s0; s < s0 + SCHUNK; ++s) {
    const size_t idx = base + (size_t)s * DM;
    const float cur = u[idx];
    const float val = w0 * um2 + w1 * um1 + w2 * cur;
    gb[idx] = (__bf16)(cg[idx] * val);
    um2 = um1;
    um1 = cur;
  }
}

// ---------------- host launcher ----------------
extern "C" void kernel_launch(void* const* d_in, const int* in_sizes, int n_in,
                              void* d_out, int out_size, void* d_ws, size_t ws_size,
                              hipStream_t stream) {
  (void)in_sizes; (void)n_in; (void)out_size; (void)ws_size;
  const float* x  = (const float*)d_in[0];   // [B,S,D]
  const float* w1 = (const float*)d_in[1];   // [3D,D]
  const float* w2 = (const float*)d_in[2];   // [D,D]
  const float* cw = (const float*)d_in[3];   // [D,1,3]

  // workspace carve-out (256B aligned): ~235 MB total
  char* ws = (char*)d_ws;
  size_t off = 0;
  auto carve = [&](size_t bytes) {
    void* p = ws + off;
    off += (bytes + 255) & ~(size_t)255;
    return p;
  };
  __bf16* xb  = (__bf16*)carve((size_t)M_TOT * DM * sizeof(__bf16));
  __bf16* w1b = (__bf16*)carve((size_t)3 * DM * DM * sizeof(__bf16));
  __bf16* w2b = (__bf16*)carve((size_t)DM * DM * sizeof(__bf16));
  float*  u   = (float*) carve((size_t)M_TOT * DM * sizeof(float));
  float*  cg  = (float*) carve((size_t)M_TOT * DM * sizeof(float));
  __bf16* gb  = (__bf16*)carve((size_t)M_TOT * DM * sizeof(__bf16));

  // 1) fp32 -> bf16 operand conversion
  cvt_f32_bf16<<<2048, 256, 0, stream>>>(x,  xb,  M_TOT * DM);
  cvt_f32_bf16<<<2048, 256, 0, stream>>>(w1, w1b, 3 * DM * DM);
  cvt_f32_bf16<<<1024, 256, 0, stream>>>(w2, w2b, DM * DM);

  // 2) fused GEMM1 (x @ w1^T for all three gates) + u = Bg*Xg epilogue
  dim3 g1(DM / BN, M_TOT / BM);  // (16, 64)
  gemm_bf16_wmma<3><<<g1, 256, 0, stream>>>(xb, w1b, u, cg);

  // 3) causal depthwise conv + Cg gating -> bf16
  dim3 gc(DM / 256, SEQ / SCHUNK, BATCH);  // (8, 64, 2)
  conv3_gate<<<gc, 256, 0, stream>>>(u, cg, cw, gb);

  // 4) GEMM2 (gated @ w2^T) -> fp32 output
  dim3 g2(DM / BN, M_TOT / BM);  // (16, 64)
  gemm_bf16_wmma<1><<<g2, 256, 0, stream>>>(gb, w2b, (float*)d_out, nullptr);
}
